// LSTMDecoder_9311489097787
// MI455X (gfx1250) — matrix operand order repaired and checked by reading, not compiled, not run on hardware
//
#include <hip/hip_runtime.h>
#include <hip/hip_bf16.h>

// Problem constants (match reference)
#define C_IN 95
#define CP   96      // K padded to 96 = 3 chunks of 32
#define H_   128
#define G4   512     // 4*H
#define B_   256
#define T_   1000
#define O_   4
#define EPS_ 1e-5f

typedef __attribute__((ext_vector_type(16))) __bf16 v16bf;
typedef __attribute__((ext_vector_type(8)))  float  v8f;

union BFrag { v16bf v; uint4 q[2]; };
union FFrag { v8f  v; float f[8]; };

__device__ __forceinline__ __bf16 f2bf(float f) {
    unsigned u = __float_as_uint(f);
    unsigned r = u + 0x7FFFu + ((u >> 16) & 1u);   // round-to-nearest-even
    unsigned short h = (unsigned short)(r >> 16);
    __bf16 out;
    __builtin_memcpy(&out, &h, 2);
    return out;
}
__device__ __forceinline__ float bf2f(__bf16 b) {
    unsigned short h;
    __builtin_memcpy(&h, &b, 2);
    return __uint_as_float(((unsigned)h) << 16);
}
__device__ __forceinline__ float sigm_(float x) { return 1.0f / (1.0f + __expf(-x)); }
__device__ __forceinline__ float tanh_(float x) {
    float e = __expf(-2.0f * x);
    return (1.0f - e) / (1.0f + e);
}

__device__ __forceinline__ v8f wmma_bf16(v16bf a, v16bf b, v8f c) {
    // (neg_a, A, neg_b, B, c_mod, C, reuse_a, reuse_b)
    return __builtin_amdgcn_wmma_f32_16x16x32_bf16(false, a, false, b, (short)0, c,
                                                   false, false);
}

// A fragment (16x32 bf16) from LDS, rows row-major, K contiguous (strideK bf16/row).
// Lane L (L<16): row=L, K = kc*32 + {0..7, 16..23}; L>=16: row=L-16, K offset +8.
__device__ __forceinline__ v16bf load_a_frag(const __bf16* base, int strideK, int kc, int lane) {
    int row  = lane & 15;
    int half = lane >> 4;
    const __bf16* p = base + row * strideK + kc * 32 + half * 8;
    BFrag fr;
    fr.q[0] = *(const uint4*)(p);        // K = kc*32 + half*8 + [0..7]
    fr.q[1] = *(const uint4*)(p + 16);   // K = kc*32 + 16 + half*8 + [0..7]
    return fr.v;
}

// B fragment (32x16 bf16) from LDS where column n is K-contiguous (lds[n*strideK + k]).
// Lane L (L<16): col=L, K = kc*32 + {0..7, 16..23}; L>=16: col=L-16, K offset +8.
__device__ __forceinline__ v16bf load_b_frag(const __bf16* base, int strideK, int n0, int kc, int lane) {
    int col = n0 + (lane & 15);
    int kb  = kc * 32 + (lane >> 4) * 8;
    const __bf16* p = base + col * strideK + kb;
    BFrag fr;
    fr.q[0] = *(const uint4*)(p);
    fr.q[1] = *(const uint4*)(p + 16);
    return fr.v;
}

// ---------------------------------------------------------------------------
// Kernel 1: xW[t][b][g] = BN(x)[b][t][:] @ W_ih^T + (b_ih + b_hh)
// Output layout [T, B, 512] so the scan kernel reads contiguous blocks per step.
// Grid: (B*T)/128 = 2000 blocks of 256 threads (8 waves, 1 M-tile each).
// ---------------------------------------------------------------------------
__global__ __launch_bounds__(256) void k_precompute_xw(
    const float* __restrict__ x,     const float* __restrict__ gamma,
    const float* __restrict__ beta,  const float* __restrict__ rmean,
    const float* __restrict__ rvar,  const float* __restrict__ Wih,
    const float* __restrict__ bih,   const float* __restrict__ bhh,
    float* __restrict__ xW)
{
    extern __shared__ char smem[];
    __bf16* sWih   = (__bf16*)(smem);             // [512][96] bf16 = 98304 B
    __bf16* sXn    = (__bf16*)(smem + 98304);     // [128][96] bf16 = 24576 B
    float*  sBias  = (float*)(smem + 122880);     // [512] f32
    float*  sScale = (float*)(smem + 124928);     // [96]  f32
    float*  sShift = (float*)(smem + 125312);     // [96]  f32

    const int tid = threadIdx.x;

    // BN affine fold: xn = x*scale + shift
    if (tid < CP) {
        float sc = 0.f, sh = 0.f;
        if (tid < C_IN) {
            sc = gamma[tid] * rsqrtf(rvar[tid] + EPS_);
            sh = beta[tid] - rmean[tid] * sc;
        }
        sScale[tid] = sc; sShift[tid] = sh;
    }
    for (int n = tid; n < G4; n += 256) sBias[n] = bih[n] + bhh[n];
    // W_ih rows -> LDS bf16 (K-contiguous per gate row; pad k==95 with 0)
    for (int idx = tid; idx < G4 * CP; idx += 256) {
        int n = idx / CP, k = idx - n * CP;
        sWih[idx] = f2bf((k < C_IN) ? Wih[n * C_IN + k] : 0.f);
    }
    __syncthreads();

    // Stage 128 normalized input rows (flattened m = b*T + t)
    const long m0 = (long)blockIdx.x * 128;
    for (int idx = tid; idx < 128 * CP; idx += 256) {
        int r = idx / CP, k = idx - r * CP;
        float v = (k < C_IN) ? x[(m0 + r) * C_IN + k] : 0.f;
        sXn[idx] = f2bf(v * sScale[k] + sShift[k]);
    }
    __syncthreads();

    const int wave = tid >> 5, lane = tid & 31;
    const int col = lane & 15, rowHalf = (lane >> 4) * 8;
    const __bf16* aBase = sXn + wave * 16 * CP;
    v16bf a0 = load_a_frag(aBase, CP, 0, lane);
    v16bf a1 = load_a_frag(aBase, CP, 1, lane);
    v16bf a2 = load_a_frag(aBase, CP, 2, lane);

    // Hoist the [b*T+t] -> [t*B+b] transpose addressing out of the tile loop:
    // 8 output row base pointers computed ONCE (8 div/mods total per lane).
    const long mBase = m0 + wave * 16 + rowHalf;
    float* rowPtr[8];
#pragma unroll
    for (int r = 0; r < 8; ++r) {
        long m = mBase + r;                    // m = b*T + t
        long b = m / T_;
        long t = m - b * T_;
        rowPtr[r] = xW + (t * B_ + b) * G4 + col;
    }

    for (int nt = 0; nt < 32; ++nt) {
        const int n0 = nt * 16;
        v16bf b0 = load_b_frag(sWih, CP, n0, 0, lane);
        v16bf b1 = load_b_frag(sWih, CP, n0, 1, lane);
        v16bf b2 = load_b_frag(sWih, CP, n0, 2, lane);
        FFrag acc;
#pragma unroll
        for (int r = 0; r < 8; ++r) acc.f[r] = 0.f;
        acc.v = wmma_bf16(a0, b0, acc.v);
        acc.v = wmma_bf16(a1, b1, acc.v);
        acc.v = wmma_bf16(a2, b2, acc.v);

        const float bb = sBias[n0 + col];
#pragma unroll
        for (int r = 0; r < 8; ++r)
            rowPtr[r][n0] = acc.f[r] + bb;     // simple add + global_store_b32
    }
}

// ---------------------------------------------------------------------------
// Kernel 2: sequential LSTM scan + final FC.
// Grid: 16 blocks (batch tiles of 16) x 512 threads (16 waves).
// Each wave owns 32 gate columns; its 8 W_hh B-fragments live in VGPRs for
// the whole 1000-step loop. h (bf16) / c (f32) / activated gates live in LDS.
// ---------------------------------------------------------------------------
__global__ __launch_bounds__(512) void k_lstm_scan(
    const float* __restrict__ Whh, const float* __restrict__ xW,
    const float* __restrict__ fcw, const float* __restrict__ fcb,
    float* __restrict__ out)
{
    extern __shared__ char smem[];
    __bf16* sWhh = (__bf16*)(smem);            // phase 1: [512][128] bf16 = 131072 B
    float*  sG   = (float*)(smem);             // phase 2: gates [16][512] f32 = 32 KB
    float*  sC   = (float*)(smem + 32768);     // c state [16][128] f32
    __bf16* sH   = (__bf16*)(smem + 40960);    // h state [16][128] bf16

    const int tid = threadIdx.x, lane = tid & 31, wave = tid >> 5;

    // Stage W_hh as bf16, K-contiguous per gate row
    for (int idx = tid; idx < G4 * H_; idx += 512) sWhh[idx] = f2bf(Whh[idx]);
    __syncthreads();

    // Hoist this wave's 8 B fragments (2 N-tiles x 4 K-chunks) into registers
    const int n0 = wave * 32;
    v16bf Bf[2][4];
#pragma unroll
    for (int nt = 0; nt < 2; ++nt)
#pragma unroll
        for (int kc = 0; kc < 4; ++kc)
            Bf[nt][kc] = load_b_frag(sWhh, H_, n0 + nt * 16, kc, lane);
    __syncthreads();   // W_hh LDS region is now dead; reuse it for gates/state

    for (int idx = tid; idx < 16 * H_; idx += 512) { sC[idx] = 0.f; sH[idx] = f2bf(0.f); }
    __syncthreads();

    const int  b0 = blockIdx.x * 16;
    const int  col = lane & 15, rowHalf = (lane >> 4) * 8;
    const int  gate = wave >> 2;               // 0=i 1=f 2=g 3=o
    const int  colOff = n0 + col;

    // Strength-reduced xW addressing: one base pointer per lane, advanced by
    // B*512 floats per step; row offsets r*512 become immediate IOFFSETs.
    const float* xwPtr = xW + (long)(b0 + rowHalf) * G4 + colOff;
    const long   stepStride = (long)B_ * G4;   // 131072 floats / step
    float*       sGlane = sG + rowHalf * G4 + colOff;

    for (int t = 0; t < T_; ++t) {
        if (t + 1 < T_)                        // prefetch next step's xW block
            __builtin_prefetch(xwPtr + stepStride, 0, 1);   // global_prefetch_b8

        v16bf a0 = load_a_frag(sH, H_, 0, lane);
        v16bf a1 = load_a_frag(sH, H_, 1, lane);
        v16bf a2 = load_a_frag(sH, H_, 2, lane);
        v16bf a3 = load_a_frag(sH, H_, 3, lane);

#pragma unroll
        for (int nt = 0; nt < 2; ++nt) {
            FFrag acc;
#pragma unroll
            for (int r = 0; r < 8; ++r)        // contiguous [T,B,512] reads
                acc.f[r] = xwPtr[r * G4 + nt * 16];
            acc.v = wmma_bf16(a0, Bf[nt][0], acc.v);
            acc.v = wmma_bf16(a1, Bf[nt][1], acc.v);
            acc.v = wmma_bf16(a2, Bf[nt][2], acc.v);
            acc.v = wmma_bf16(a3, Bf[nt][3], acc.v);
#pragma unroll
            for (int r = 0; r < 8; ++r) {
                float v = acc.f[r];
                v = (gate == 2) ? tanh_(v) : sigm_(v);
                sGlane[r * G4 + nt * 16] = v;  // immediate DS offsets
            }
        }
        __syncthreads();

        // Elementwise c/h update: 2048 elems / 512 threads = 4 each
#pragma unroll
        for (int e = 0; e < 4; ++e) {
            int flat = tid + e * 512;          // = br*128 + j
            int br = flat >> 7, j = flat & 127;
            float gi = sG[br * G4 + j];
            float gf = sG[br * G4 + 128 + j];
            float gg = sG[br * G4 + 256 + j];
            float go = sG[br * G4 + 384 + j];
            float cN = gf * sC[flat] + gi * gg;
            sC[flat] = cN;
            sH[flat] = f2bf(go * tanh_(cN));
        }
        __syncthreads();

        xwPtr += stepStride;
    }

    // Final FC: out[b0+br][oi] = h_T . fc_w[oi] + fc_b[oi]
    if (tid < 16 * O_) {
        int br = tid >> 2, oi = tid & 3;
        float s = fcb[oi];
        for (int k = 0; k < H_; ++k)
            s += bf2f(sH[br * H_ + k]) * fcw[oi * H_ + k];
        out[(b0 + br) * O_ + oi] = s;
    }
}

// ---------------------------------------------------------------------------
extern "C" void kernel_launch(void* const* d_in, const int* in_sizes, int n_in,
                              void* d_out, int out_size, void* d_ws, size_t ws_size,
                              hipStream_t stream) {
    const float* x     = (const float*)d_in[0];
    const float* gamma = (const float*)d_in[1];
    const float* beta  = (const float*)d_in[2];
    const float* rmean = (const float*)d_in[3];
    const float* rvar  = (const float*)d_in[4];
    const float* Wih   = (const float*)d_in[5];
    const float* Whh   = (const float*)d_in[6];
    const float* bih   = (const float*)d_in[7];
    const float* bhh   = (const float*)d_in[8];
    const float* fcw   = (const float*)d_in[9];
    const float* fcb   = (const float*)d_in[10];

    float* xW = (float*)d_ws;   // [T, B, 512] f32 = 524,288,000 bytes

    const size_t lds1 = 125696;   // Wih(96K) + xn(24K) + bias/bn coeffs
    const size_t lds2 = 131072;   // Whh staging, reused as gates/c/h

    k_precompute_xw<<<(B_ * T_) / 128, 256, lds1, stream>>>(
        x, gamma, beta, rmean, rvar, Wih, bih, bhh, xW);
    k_lstm_scan<<<B_ / 16, 512, lds2, stream>>>(Whh, xW, fcw, fcb, (float*)d_out);
}
